// RNNModel_50053548867903
// MI455X (gfx1250) — matrix operand order repaired and checked by reading, not compiled, not run on hardware
//
#include <hip/hip_runtime.h>

typedef _Float16 f16;
typedef __attribute__((ext_vector_type(16))) _Float16 v16h;
typedef __attribute__((ext_vector_type(8)))  float   v8f;

#define B_ 256
#define T_ 2048
#define F_ 2
#define H_ 128
#define O_ 2
#define LDSROW 136   // halves per LDS row: 128 + 8 pad -> 272B stride (4-bank rotation, conflict-free b128 reads)

union V16U { v16h v; float4 q[2]; f16 h[16]; };

#if __has_builtin(__builtin_amdgcn_exp2f)
#define EXP2F(x) __builtin_amdgcn_exp2f(x)
#else
#define EXP2F(x) exp2f(x)
#endif
#if __has_builtin(__builtin_amdgcn_rcpf)
#define RCPF(x) __builtin_amdgcn_rcpf(x)
#else
#define RCPF(x) (1.0f / (x))
#endif

__device__ __forceinline__ float sig_f(float x) {
  // 1/(1+2^(-x*log2 e)) -> v_exp_f32 + v_rcp_f32
  return RCPF(1.0f + EXP2F(-1.44269504f * x));
}
__device__ __forceinline__ float tanh_f(float x) {
  // tanh(x) = 2*sigmoid(2x)-1
  return 2.0f * RCPF(1.0f + EXP2F(-2.88539008f * x)) - 1.0f;
}

#define WMMA_F16(A, Bm, C) \
  __builtin_amdgcn_wmma_f32_16x16x32_f16(false, (A), false, (Bm), (short)0, (C), false, false)

__global__ __launch_bounds__(256)
void gru_scan_kernel(const float* __restrict__ x,
                     const float* __restrict__ h0,
                     const float* __restrict__ Wir, const float* __restrict__ Wiz,
                     const float* __restrict__ Win,
                     const float* __restrict__ Whr, const float* __restrict__ Whz,
                     const float* __restrict__ Whn,
                     const float* __restrict__ bhr, const float* __restrict__ bhz,
                     const float* __restrict__ bhn,
                     const float* __restrict__ Wd,  const float* __restrict__ bd,
                     float* __restrict__ carry_out, float* __restrict__ out)
{
  __shared__ f16   hbuf[2][16 * LDSROW];      // double-buffered h tile (f16)
  __shared__ float scr[2][O_][16][8];          // out-projection partials

  const int  tid    = threadIdx.x;
  const int  lane   = tid & 31;
  const int  w      = tid >> 5;                // wave id == N-tile (16 cols of H)
  const int  b0     = blockIdx.x * 16;         // batch tile base
  const int  row    = lane & 15;               // A row (M) / B,C column (N) within tile
  const bool hiHalf = (lane >= 16);
  const int  nglob  = w * 16 + row;            // global hidden index of this lane's column
  const int  mofs   = hiHalf ? 8 : 0;          // C/D: element e -> M = e + mofs
  const int  kofsA  = hiHalf ? 8 : 0;          // A f16 layout K offset per lane half
  const int  kofsB  = hiHalf ? 16 : 0;         // B f16 layout K offset per lane half

  // ---------- one-time: Wh* -> WMMA B-operand registers (f16) ----------
  // B 32x16 f16 layout: lane = N (mod 16); halves m=0..7 hold K = kofsB/2 + m, m=8..15 hold K = 16 + kofsB/2 + (m-8).
  v16h br[5], bz[5], bn4[4], bxn;
  #pragma unroll
  for (int c = 0; c < 4; ++c) {
    V16U ur, uz, un;
    #pragma unroll
    for (int m = 0; m < 16; ++m) {
      const int K = 32 * c + (kofsB >> 1) + (m < 8 ? m : m + 8);
      ur.h[m] = (f16)Whr[K * H_ + nglob];
      uz.h[m] = (f16)Whz[K * H_ + nglob];
      un.h[m] = (f16)Whn[K * H_ + nglob];
    }
    br[c] = ur.v; bz[c] = uz.v; bn4[c] = un.v;
  }
  { // x-chunk B operands: only rows K=0,1 nonzero (F=2), rest zero
    V16U ur, uz, un;
    #pragma unroll
    for (int m = 0; m < 16; ++m) { ur.h[m] = (f16)0.f; uz.h[m] = (f16)0.f; un.h[m] = (f16)0.f; }
    if (!hiHalf) {
      ur.h[0] = (f16)Wir[0 * H_ + nglob]; ur.h[1] = (f16)Wir[1 * H_ + nglob];
      uz.h[0] = (f16)Wiz[0 * H_ + nglob]; uz.h[1] = (f16)Wiz[1 * H_ + nglob];
      un.h[0] = (f16)Win[0 * H_ + nglob]; un.h[1] = (f16)Win[1 * H_ + nglob];
    }
    br[4] = ur.v; bz[4] = uz.v; bxn = un.v;
  }

  // loop-invariant accumulator seeds (biases broadcast across the 8 M rows)
  const float biasr = bhr[nglob], biasz = bhz[nglob], biasn = bhn[nglob];
  v8f crinit, czinit, cninit, zinit;
  #pragma unroll
  for (int e = 0; e < 8; ++e) {
    crinit[e] = biasr; czinit[e] = biasz; cninit[e] = biasn; zinit[e] = 0.f;
  }

  // out-projection role of this thread: row pr_r, output pr_o, k-segment pr_seg
  const int pr_r = tid & 15, pr_o = (tid >> 4) & 1, pr_seg = tid >> 5;
  float wdreg[16];
  #pragma unroll
  for (int q = 0; q < 16; ++q) wdreg[q] = Wd[(pr_seg * 16 + q) * O_ + pr_o];
  const float bdv = bd[pr_o];
  const int prOfs = pr_r * LDSROW + pr_seg * 16;   // halves offset for stage-1 reads

  // hprev fp32 tile in registers (element e -> (b0+e+mofs, nglob))
  v8f hprev;
  #pragma unroll
  for (int e = 0; e < 8; ++e) hprev[e] = h0[(size_t)(b0 + e + mofs) * H_ + nglob];

  // seed hbuf[0] with h0 as f16
  #pragma unroll
  for (int e = 0; e < 8; ++e)
    hbuf[0][(e + mofs) * LDSROW + nglob] = (f16)hprev[e];

  // x prefetch for t=0 (each lane: its row's F=2 feature pair)
  const float* xrowp = x + (size_t)(b0 + row) * T_ * F_;
  float2 xcur = *(const float2*)(xrowp);

  // persistent x-chunk A operand: only halves 0,1 change per step
  V16U ua5;
  #pragma unroll
  for (int m = 0; m < 16; ++m) ua5.h[m] = (f16)0.f;

  const int aOfs = row * LDSROW + kofsA;           // halves offset for A-rebuild reads

  __syncthreads();

  for (int t = 0; t < T_; ++t) {
    const int  p  = t & 1;
    const f16* hb = &hbuf[p][0];           // holds h_{t-1}

    // ---- rebuild A operand chunks (h_{t-1} as f16) from LDS ----
    // A 16x32 f16 layout: lane = M (mod 16); halves 0..7 -> K = 32c+kofsA+m,
    // halves 8..15 -> K = 32c+16+kofsA+(m-8). Two b128 LDS loads per chunk.
    v16h a[4];
    #pragma unroll
    for (int c = 0; c < 4; ++c) {
      V16U ua;
      ua.q[0] = *(const float4*)(hb + aOfs + 32 * c);
      ua.q[1] = *(const float4*)(hb + aOfs + 32 * c + 16);
      a[c] = ua.v;
    }
    // A x-chunk: halves 0,1 = x_t features (only K=0,1 meet nonzero B rows)
    ua5.h[0] = (f16)xcur.x; ua5.h[1] = (f16)xcur.y;
    const v16h a5 = ua5.v;

    // prefetch x for next step (latency hidden behind WMMA/gate math)
    const int tn = (t + 1 < T_) ? (t + 1) : (T_ - 1);
    const float2 xnxt = *(const float2*)(xrowp + (size_t)tn * F_);

    // ---- out-projection pipeline stage 1: partials of ys_{t-1} ----
    if (t >= 1) {
      V16U u0;
      u0.q[0] = *(const float4*)(hb + prOfs);
      u0.q[1] = *(const float4*)(hb + prOfs + 8);
      float s = 0.f;
      #pragma unroll
      for (int q = 0; q < 16; ++q) s += (float)u0.h[q] * wdreg[q];
      scr[p][pr_o][pr_r][pr_seg] = s;
    }
    // ---- stage 2: reduce partials of ys_{t-2}, store out ----
    if (t >= 2 && tid < 32) {
      float s = bdv;
      #pragma unroll
      for (int q = 0; q < 8; ++q) s += scr[1 - p][pr_o][pr_r][q];
      out[((size_t)(b0 + pr_r) * T_ + (t - 2)) * O_ + pr_o] = s;
    }

    // ---- gate GEMMs: D = A x B + C, C seeded with loop-invariant bias vectors ----
    v8f cr = WMMA_F16(a[0], br[0],  crinit);
    v8f cz = WMMA_F16(a[0], bz[0],  czinit);
    v8f cn = WMMA_F16(a[0], bn4[0], cninit);
    #pragma unroll
    for (int c = 1; c < 4; ++c) {
      cr = WMMA_F16(a[c], br[c],  cr);
      cz = WMMA_F16(a[c], bz[c],  cz);
      cn = WMMA_F16(a[c], bn4[c], cn);
    }
    cr = WMMA_F16(a5, br[4], cr);
    cz = WMMA_F16(a5, bz[4], cz);
    v8f cx = WMMA_F16(a5, bxn, zinit);

    // ---- GRU nonlinearity + state update (fp32) ----
    v8f hnew;
    #pragma unroll
    for (int e = 0; e < 8; ++e) {
      const float r = sig_f(cr[e]);
      const float z = sig_f(cz[e]);
      const float n = tanh_f(cx[e] + r * cn[e]);   // n = tanh(xn + r*(h@Whn + bhn))
      hnew[e] = n + z * (hprev[e] - n);            // (1-z)*n + z*h
    }
    hprev = hnew;

    // publish h_t as f16 into the other buffer
    f16* hw = &hbuf[1 - p][0];
    #pragma unroll
    for (int e = 0; e < 8; ++e)
      hw[(e + mofs) * LDSROW + nglob] = (f16)hnew[e];

    xcur = xnxt;
    __syncthreads();
  }

  // ---- epilogue: drain out-projection pipeline for t = T-2, T-1 ----
  {
    const int  p  = T_ & 1;                 // hbuf[p] holds h_{T-1}
    const f16* hb = &hbuf[p][0];
    V16U u0;
    u0.q[0] = *(const float4*)(hb + prOfs);
    u0.q[1] = *(const float4*)(hb + prOfs + 8);
    float s = 0.f;
    #pragma unroll
    for (int q = 0; q < 16; ++q) s += (float)u0.h[q] * wdreg[q];
    scr[p][pr_o][pr_r][pr_seg] = s;
    if (tid < 32) {
      float s2 = bdv;
      #pragma unroll
      for (int q = 0; q < 8; ++q) s2 += scr[1 - p][pr_o][pr_r][q];
      out[((size_t)(b0 + pr_r) * T_ + (T_ - 2)) * O_ + pr_o] = s2;
    }
    __syncthreads();
    if (tid < 32) {
      float s2 = bdv;
      #pragma unroll
      for (int q = 0; q < 8; ++q) s2 += scr[p][pr_o][pr_r][q];
      out[((size_t)(b0 + pr_r) * T_ + (T_ - 1)) * O_ + pr_o] = s2;
    }
  }

  // ---- carry = h_{T-1} (fp32, straight from registers) ----
  #pragma unroll
  for (int e = 0; e < 8; ++e)
    carry_out[(size_t)(b0 + e + mofs) * H_ + nglob] = hprev[e];
}

extern "C" void kernel_launch(void* const* d_in, const int* in_sizes, int n_in,
                              void* d_out, int out_size, void* d_ws, size_t ws_size,
                              hipStream_t stream) {
  (void)in_sizes; (void)n_in; (void)d_ws; (void)ws_size; (void)out_size;
  const float* x   = (const float*)d_in[0];
  const float* h0  = (const float*)d_in[1];
  const float* Wir = (const float*)d_in[2];
  const float* Wiz = (const float*)d_in[3];
  const float* Win = (const float*)d_in[4];
  const float* Whr = (const float*)d_in[5];
  const float* Whz = (const float*)d_in[6];
  const float* Whn = (const float*)d_in[7];
  const float* bhr = (const float*)d_in[8];
  const float* bhz = (const float*)d_in[9];
  const float* bhn = (const float*)d_in[10];
  const float* Wd  = (const float*)d_in[11];
  const float* bd  = (const float*)d_in[12];

  float* carry = (float*)d_out;              // [B,H]
  float* out   = (float*)d_out + B_ * H_;    // [B,T,O]

  hipLaunchKernelGGL(gru_scan_kernel, dim3(B_ / 16), dim3(256), 0, stream,
                     x, h0, Wir, Wiz, Win, Whr, Whz, Whn, bhr, bhz, bhn, Wd, bd,
                     carry, out);
}
